// ODEFunc_39754217292311
// MI455X (gfx1250) — matrix operand order repaired and checked by reading, not compiled, not run on hardware
//
#include <hip/hip_runtime.h>

typedef __attribute__((ext_vector_type(16))) _Float16 v16h;
typedef __attribute__((ext_vector_type(8)))  _Float16 v8h;
typedef __attribute__((ext_vector_type(8)))  float    v8f;
typedef _Float16 half_t;

#define SD   10
#define HID  128
#define SE   16
#define NC   12
#define KXS  26   // sd + se
#define KFE  21   // 5 + se
#define EH   64   // hd/2

// Fragment pool layout in d_ws (each fragment: 32 lanes x 16 halves = 1KB)
#define FRAG_W1_OFF   0   // 8 frags: nt = 0..7          (K=26 -> 1 ktile)
#define FRAG_W2_OFF   8   // 32 frags: kt*8 + nt
#define FRAG_W3_OFF   40  // 4 frags: kt = 0..3          (N=10 -> 1 ntile)
#define FRAG_WLIN_OFF 44  // 1 frag
#define FRAG_WE1_OFF  45  // 4 frags: nt = 0..3
#define NFRAGS        49

__device__ __forceinline__ float fast_tanh(float v) {
#if __has_builtin(__builtin_amdgcn_tanhf)
  return __builtin_amdgcn_tanhf(v);
#else
  return tanhf(v);
#endif
}

// ---------------------------------------------------------------------------
// Pack all weight matrices (f32, row-major KxN) into WMMA B-fragment layout
// (f16).  B-layout (16x16x32 f16): slot j of lane l holds
//   W[kt*32 + (l/16)*8 + j (j<8) | kt*32 + 16 + (l/16)*8 + (j-8) (j>=8)][nt*16 + (l%16)]   (zero outside the matrix).
// One block (32 threads) per fragment.
// ---------------------------------------------------------------------------
__global__ void pack_wmma_frags(const float* __restrict__ W1,
                                const float* __restrict__ W2,
                                const float* __restrict__ W3,
                                const float* __restrict__ W_lin,
                                const float* __restrict__ We1,
                                half_t* __restrict__ frags) {
  int f = blockIdx.x;
  int lane = threadIdx.x;  // 0..31
  const float* W; int K, N, kt, nt;
  if (f < 8)       { W = W1;    K = KXS; N = HID; kt = 0;         nt = f;       }
  else if (f < 40) { W = W2;    K = HID; N = HID; kt = (f - 8)/8; nt = (f-8)%8; }
  else if (f < 44) { W = W3;    K = HID; N = SD;  kt = f - 40;    nt = 0;       }
  else if (f < 45) { W = W_lin; K = KXS; N = SD;  kt = 0;         nt = 0;       }
  else             { W = We1;   K = KFE; N = EH;  kt = 0;         nt = f - 45;  }

  half_t* dst = frags + (size_t)f * 512 + (size_t)lane * 16;
  int kbase = kt * 32 + (lane >> 4) * 8;
  int n     = nt * 16 + (lane & 15);
#pragma unroll
  for (int j = 0; j < 16; ++j) {
    int k = kbase + (j < 8 ? j : 16 + (j - 8));
    float v = (k < K && n < N) ? W[k * N + n] : 0.0f;
    dst[j] = (half_t)v;
  }
}

// ---------------------------------------------------------------------------
// Main fused ODE kernel: one 16-row batch tile per wave, 8 waves per block.
// 49 v_wmma_f32_16x16x32_f16 per tile.
// ---------------------------------------------------------------------------
__global__ void __launch_bounds__(256)
odefunc_wmma(const float* __restrict__ t,
             const float* __restrict__ x,
             const float* __restrict__ emb,
             const float* __restrict__ b_lin,
             const float* __restrict__ b1,
             const float* __restrict__ b2,
             const float* __restrict__ b3,
             const float* __restrict__ be1,
             const float* __restrict__ We2,
             const float* __restrict__ be2,
             const half_t* __restrict__ frags,
             float* __restrict__ out,
             int B, int tiles) {
  // per-wave private h staging: [m][k], row pitch 144 halves (288B, 16B aligned,
  // bank-conflict-free for the b128 A-fragment reads)
  __shared__ half_t hbuf[8][16][144];

  const int lane = threadIdx.x & 31;
  const int w    = threadIdx.x >> 5;
  const int ml   = lane & 15;   // A-frag row / B-C-frag column
  const int hi   = lane >> 4;

  // --- seasonal embedding (wave-uniform), kept packed in one v16h (8 VGPRs) ---
  float tv  = t[0];
  int   idx = (int)(tv * (float)NC);
  idx %= NC; if (idx < 0) idx += NC;
  v16h sev;
#pragma unroll
  for (int j = 0; j < SE; ++j) sev[j] = (half_t)emb[idx * SE + j];

  // --- per-lane column-dependent constants (loop invariant) ---
  float b1v[8], b2v[8];
#pragma unroll
  for (int nt = 0; nt < 8; ++nt) { b1v[nt] = b1[nt*16 + ml]; b2v[nt] = b2[nt*16 + ml]; }
  float b3v   = (ml < SD) ? b3[ml]    : 0.0f;
  float blinv = (ml < SD) ? b_lin[ml] : 0.0f;
  float be1v[4], w2c0[4], w2c1[4];
#pragma unroll
  for (int nt = 0; nt < 4; ++nt) {
    be1v[nt] = be1[nt*16 + ml];
    w2c0[nt] = We2[(nt*16 + ml)*2 + 0];
    w2c1[nt] = We2[(nt*16 + ml)*2 + 1];
  }
  const float be2_0 = be2[0], be2_1 = be2[1];

  const v16h* __restrict__ vfrags = (const v16h*)frags;
  const int gw = blockIdx.x * 8 + w;
  const int nw = gridDim.x * 8;
  const half_t hz = (half_t)0.0f;

#pragma clang loop unroll(disable)
  for (int tile = gw; tile < tiles; tile += nw) {
    // Launder a zero *index* through an empty asm each iteration: keeps the
    // fragment loads rooted at the kernarg pointer (so they stay
    // global_load_b128, LOADcnt-only), but makes their addresses loop-variant
    // so LICM cannot hoist 136+ VGPRs of weights out of the loop.
    unsigned zoff = 0;
    asm volatile("" : "+s"(zoff));
    const v16h* vfr = vfrags + zoff;

    const long tbase = (long)tile * 16;
    long rowA = tbase + ml; if (rowA >= B) rowA = (long)B - 1;
    const float* xr = x + rowA * SD;

    // prefetch next tile's x rows
    long nxt = (long)(tile + nw) * 16 + ml;
    if (nxt < B) __builtin_prefetch(x + nxt * SD, 0, 1);

    // --- load x row pieces (contiguous, 8B aligned) ---
    float xv0=0,xv1=0,xv2=0,xv3=0,xv4=0,xv5=0,xv6=0,xv7=0;
    if (hi == 0) {
      float2 p0 = *(const float2*)(xr + 0);
      float2 p1 = *(const float2*)(xr + 2);
      float2 p2 = *(const float2*)(xr + 4);
      float2 p3 = *(const float2*)(xr + 6);
      xv0=p0.x; xv1=p0.y; xv2=p1.x; xv3=p1.y;
      xv4=p2.x; xv5=p2.y; xv6=p3.x; xv7=p3.y;
    } else {
      float2 p4 = *(const float2*)(xr + 8);
      xv0=p4.x; xv1=p4.y;
    }

    // --- A fragment: xs = [x(10) | se(16) | 0-pad(6)], 16x32 f16 ---
    v16h axs;
    if (hi == 0) {             // k = 0..7, 16..23
      axs[0]=(half_t)xv0; axs[1]=(half_t)xv1; axs[2]=(half_t)xv2; axs[3]=(half_t)xv3;
      axs[4]=(half_t)xv4; axs[5]=(half_t)xv5; axs[6]=(half_t)xv6; axs[7]=(half_t)xv7;
      axs[8]=sev[6];  axs[9]=sev[7];  axs[10]=sev[8];  axs[11]=sev[9];
      axs[12]=sev[10];axs[13]=sev[11];axs[14]=sev[12]; axs[15]=sev[13];
    } else {                   // k = 8..15, 24..31
      axs[0]=(half_t)xv0; axs[1]=(half_t)xv1;
      axs[2]=sev[0]; axs[3]=sev[1]; axs[4]=sev[2];
      axs[5]=sev[3]; axs[6]=sev[4]; axs[7]=sev[5];
      axs[8]=sev[14]; axs[9]=sev[15];
      axs[10]=hz; axs[11]=hz; axs[12]=hz; axs[13]=hz; axs[14]=hz; axs[15]=hz;
    }

    // --- A fragment: fe = [T*T,T*H,T^3,T^2*H,T*H^2 | se(16) | 0-pad(11)] ---
    v16h afe;
    if (hi == 0) {             // k = 0..7, 16..23
      float T = xv0, H = xv1;
      afe[0]=(half_t)(T*T);   afe[1]=(half_t)(T*H);   afe[2]=(half_t)(T*T*T);
      afe[3]=(half_t)(T*T*H); afe[4]=(half_t)(T*H*H);
      afe[5]=sev[0]; afe[6]=sev[1]; afe[7]=sev[2];
      afe[8]=sev[11]; afe[9]=sev[12]; afe[10]=sev[13]; afe[11]=sev[14]; afe[12]=sev[15];
      afe[13]=hz; afe[14]=hz; afe[15]=hz;
    } else {                   // k = 8..15, 24..31
      afe[0]=sev[3]; afe[1]=sev[4]; afe[2]=sev[5]; afe[3]=sev[6];
      afe[4]=sev[7]; afe[5]=sev[8]; afe[6]=sev[9]; afe[7]=sev[10];
      afe[8]=hz; afe[9]=hz; afe[10]=hz; afe[11]=hz; afe[12]=hz; afe[13]=hz; afe[14]=hz; afe[15]=hz;
    }

    // --- linear term: xs @ W_lin + b_lin (1 WMMA) ---
    v8f clin;
#pragma unroll
    for (int i = 0; i < 8; ++i) clin[i] = blinv;
    {
      v16h bw = vfr[(size_t)FRAG_WLIN_OFF * 32 + lane];
      clin = __builtin_amdgcn_wmma_f32_16x16x32_f16(false, axs, false, bw,
                                                    (short)0, clin, false, false);
    }

    // --- layer 1: h1 = tanh(xs @ W1 + b1) -> LDS [m][k] (8 WMMA) ---
#pragma unroll
    for (int nt = 0; nt < 8; ++nt) {
      v8f c;
#pragma unroll
      for (int i = 0; i < 8; ++i) c[i] = b1v[nt];
      v16h bw = vfr[(size_t)(FRAG_W1_OFF + nt) * 32 + lane];
      c = __builtin_amdgcn_wmma_f32_16x16x32_f16(false, axs, false, bw,
                                                 (short)0, c, false, false);
#pragma unroll
      for (int i = 0; i < 8; ++i)
        hbuf[w][i + 8*hi][nt*16 + ml] = (half_t)fast_tanh(c[i]);
    }

    // --- gather h1 A-fragments (same-wave DS ops are in-order) ---
    v16h a2[4];
#pragma unroll
    for (int kt = 0; kt < 4; ++kt) {
      v8h lo = *(const v8h*)&hbuf[w][ml][kt*32 + hi*8];
      v8h hu = *(const v8h*)&hbuf[w][ml][kt*32 + 16 + hi*8];
#pragma unroll
      for (int j = 0; j < 8; ++j) { a2[kt][j] = lo[j]; a2[kt][j + 8] = hu[j]; }
    }

    // --- layer 2: h2 = tanh(h1 @ W2 + b2) -> LDS (32 WMMA) ---
#pragma unroll
    for (int nt = 0; nt < 8; ++nt) {
      v8f c;
#pragma unroll
      for (int i = 0; i < 8; ++i) c[i] = b2v[nt];
#pragma unroll
      for (int kt = 0; kt < 4; ++kt) {
        v16h bw = vfr[(size_t)(FRAG_W2_OFF + kt*8 + nt) * 32 + lane];
        c = __builtin_amdgcn_wmma_f32_16x16x32_f16(false, a2[kt], false, bw,
                                                   (short)0, c, false, false);
      }
#pragma unroll
      for (int i = 0; i < 8; ++i)
        hbuf[w][i + 8*hi][nt*16 + ml] = (half_t)fast_tanh(c[i]);
    }

    // --- gather h2 A-fragments ---
    v16h a3[4];
#pragma unroll
    for (int kt = 0; kt < 4; ++kt) {
      v8h lo = *(const v8h*)&hbuf[w][ml][kt*32 + hi*8];
      v8h hu = *(const v8h*)&hbuf[w][ml][kt*32 + 16 + hi*8];
#pragma unroll
      for (int j = 0; j < 8; ++j) { a3[kt][j] = lo[j]; a3[kt][j + 8] = hu[j]; }
    }

    // --- layer 3: nonlinear = h2 @ W3 + b3 (4 WMMA) ---
    v8f c3;
#pragma unroll
    for (int i = 0; i < 8; ++i) c3[i] = b3v;
#pragma unroll
    for (int kt = 0; kt < 4; ++kt) {
      v16h bw = vfr[(size_t)(FRAG_W3_OFF + kt) * 32 + lane];
      c3 = __builtin_amdgcn_wmma_f32_16x16x32_f16(false, a3[kt], false, bw,
                                                  (short)0, c3, false, false);
    }

    // --- ENSO head: tanh(fe @ We1 + be1) @ We2 (4 WMMA + shuffle reduce) ---
    // Done last so its 16 partial-sum registers are not live across the MLP.
    float p0[8], p1[8];
#pragma unroll
    for (int i = 0; i < 8; ++i) { p0[i] = 0.0f; p1[i] = 0.0f; }
#pragma unroll
    for (int nt = 0; nt < 4; ++nt) {
      v8f ce;
#pragma unroll
      for (int i = 0; i < 8; ++i) ce[i] = be1v[nt];
      v16h bw = vfr[(size_t)(FRAG_WE1_OFF + nt) * 32 + lane];
      ce = __builtin_amdgcn_wmma_f32_16x16x32_f16(false, afe, false, bw,
                                                  (short)0, ce, false, false);
#pragma unroll
      for (int i = 0; i < 8; ++i) {
        float th = fast_tanh(ce[i]);
        p0[i] += th * w2c0[nt];
        p1[i] += th * w2c1[nt];
      }
    }
#pragma unroll
    for (int mask = 8; mask >= 1; mask >>= 1) {
#pragma unroll
      for (int i = 0; i < 8; ++i) {
        p0[i] += __shfl_xor(p0[i], mask, 16);
        p1[i] += __shfl_xor(p1[i], mask, 16);
      }
    }

    // --- combine + store (C-layout: col = ml, row = i + 8*hi) ---
    if (ml < SD) {
#pragma unroll
      for (int i = 0; i < 8; ++i) {
        long r = tbase + i + 8*hi;
        if (r < B) {
          float eo = (ml == 0) ? (p0[i] + be2_0)
                   : (ml == 1) ? (p1[i] + be2_1) : 0.0f;
          out[r * SD + ml] = c3[i] + clin[i] + eo;
        }
      }
    }
  }
}

extern "C" void kernel_launch(void* const* d_in, const int* in_sizes, int n_in,
                              void* d_out, int out_size, void* d_ws, size_t ws_size,
                              hipStream_t stream) {
  const float* t     = (const float*)d_in[0];
  const float* x     = (const float*)d_in[1];
  const float* emb   = (const float*)d_in[2];
  const float* W_lin = (const float*)d_in[3];
  const float* b_lin = (const float*)d_in[4];
  const float* W1    = (const float*)d_in[5];
  const float* b1    = (const float*)d_in[6];
  const float* W2    = (const float*)d_in[7];
  const float* b2    = (const float*)d_in[8];
  const float* W3    = (const float*)d_in[9];
  const float* b3    = (const float*)d_in[10];
  const float* We1   = (const float*)d_in[11];
  const float* be1   = (const float*)d_in[12];
  const float* We2   = (const float*)d_in[13];
  const float* be2   = (const float*)d_in[14];

  int B = in_sizes[1] / SD;
  if (B <= 0) return;
  half_t* frags = (half_t*)d_ws;   // needs NFRAGS KB of scratch

  pack_wmma_frags<<<NFRAGS, 32, 0, stream>>>(W1, W2, W3, W_lin, We1, frags);

  int tiles = (B + 15) / 16;
  int blocks = 2048;
  if (blocks * 8 > tiles) blocks = (tiles + 7) / 8;
  if (blocks < 1) blocks = 1;
  odefunc_wmma<<<blocks, 256, 0, stream>>>(t, x, emb, b_lin, b1, b2, b3, be1,
                                           We2, be2, frags, (float*)d_out,
                                           B, tiles);
}
